// ProbabilisticSparseAttention_15942918603417
// MI455X (gfx1250) — matrix operand order repaired and checked by reading, not compiled, not run on hardware
//
#include <hip/hip_runtime.h>
#include <hip/hip_bf16.h>
#include <math.h>
#include <stdint.h>

// ---------------------------------------------------------------------------
// Problem constants (from reference)
// ---------------------------------------------------------------------------
#define Bc    8
#define LQc   512
#define LKc   2048
#define Ec    512
#define Hc    8
#define HDc   64
#define TOPKc 20

typedef __bf16 bf16_t;
typedef __attribute__((ext_vector_type(16))) __bf16 v16bf;
typedef __attribute__((ext_vector_type(8)))  __bf16 v8bf;
typedef __attribute__((ext_vector_type(8)))  float  v8f;
typedef __attribute__((ext_vector_type(4)))  unsigned int u32x4;
typedef __attribute__((ext_vector_type(8)))  int i32x8;
typedef __attribute__((ext_vector_type(4)))  int i32x4;

#if __has_builtin(__builtin_amdgcn_tensor_load_to_lds)
#define PSA_HAVE_TDM 1
#else
#define PSA_HAVE_TDM 0
#endif

// ---------------------------------------------------------------------------
// WMMA helpers (CDNA5: v_wmma_f32_16x16x32_bf16, wave32)
// ---------------------------------------------------------------------------
__device__ inline v8f wmma_bf16(v16bf a, v16bf b, v8f c) {
  // 8 args: (neg_a, A, neg_b, B, c_mod, C, reuse_a, reuse_b)
  return __builtin_amdgcn_wmma_f32_16x16x32_bf16(false, a, false, b,
                                                 (short)0, c, false, false);
}

// A fragment: 16x32 (MxK) tile, row-major source with leading dim `ld`.
// Layout: lanes 0-15 -> M=lane, K 0..15; lanes 16-31 -> M=lane-16,
// K 16..31.  Two 16-byte contiguous loads per lane.
__device__ inline v16bf load_a_frag(const bf16_t* __restrict__ base, int ld, int lane) {
  const int half = lane >> 4;
  const int m    = lane & 15;
  const bf16_t* rp = base + (long)m * ld;
  v8bf lo = *reinterpret_cast<const v8bf*>(rp + half * 16);
  v8bf hi = *reinterpret_cast<const v8bf*>(rp + half * 16 + 8);
  v16bf a;
#pragma unroll
  for (int i = 0; i < 8; ++i) { a[i] = lo[i]; a[8 + i] = hi[i]; }
  return a;
}

// B fragment for  X * W^T  style GEMMs: logical B is 32x16 (KxN) with
// B[k][n] = W[n0+n][kk+k], W row-major with leading dim `ld`.
// Contiguous-k layout used here (same as load_a_frag): lanes 0-15 -> N=lane, K=0..15; lanes 16-31 -> N=lane-16, K=16..31.
// => one contiguous 32-byte load per lane.  wbase = W + n0*ld + kk.
__device__ inline v16bf load_bT_frag(const bf16_t* __restrict__ wbase, int ld, int lane) {
  const int half = lane >> 4;
  const int n    = lane & 15;
  return *reinterpret_cast<const v16bf*>(wbase + (long)n * ld + half * 16);
}

// ---------------------------------------------------------------------------
// TDM: contiguous 1D copy global -> LDS (nelem2 elements of 2 bytes).
// Builds a D# (group0/group1 per ISA 8.3/8.4): count=1, type=2("image"),
// data_size=1 (2B), tile_dim0 = tensor_dim0 = nelem2, tensor_dim1=1.
// Issued by one wave; completion via TENSORcnt.
// ---------------------------------------------------------------------------
#if PSA_HAVE_TDM
__device__ inline void psa_tdm_load_to_lds(const void* gptr, void* lptr,
                                           unsigned nelem2) {
  const unsigned long long ga = (unsigned long long)(uintptr_t)gptr;
  // Flat addresses of LDS objects map to LDS via addr[31:0] truncation.
  const unsigned lds = (unsigned)(uintptr_t)lptr;
  u32x4 g0 = {
      1u,                                              // count=1, user mode
      lds,                                             // lds_addr [63:32]
      (unsigned)(ga & 0xffffffffu),                    // global_addr lo
      ((unsigned)((ga >> 32) & 0x01ffffffu)) | 0x80000000u  // ga hi | type=2
  };
  i32x8 g1 = {
      (int)(1u << 16),                                 // data_size=1 (2 bytes)
      (int)((nelem2 & 0xffffu) << 16),                 // tensor_dim0[15:0]
      (int)(((nelem2 >> 16) & 0xffffu) | (1u << 16)),  // tensor_dim0 hi, dim1=1
      (int)((nelem2 & 0xffffu) << 16),                 // tile_dim0
      0, 0, 0, 0                                       // tile_dim1/2=0, strides
  };
  i32x4 z4 = {0, 0, 0, 0};
#if __has_include(<hip/amd_detail/amd_gfx1250_TDM.h>)
  i32x8 z8 = {0, 0, 0, 0, 0, 0, 0, 0};
  __builtin_amdgcn_tensor_load_to_lds(g0, g1, z4, z4, z8, 0);   // clang-23 form
#else
  __builtin_amdgcn_tensor_load_to_lds(g0, g1, z4, z4, 0);       // ROCm 7.2 form
#endif
}
#endif

// ---------------------------------------------------------------------------
// f32 -> bf16 conversion (grid-stride)
// ---------------------------------------------------------------------------
__global__ void psa_f32_to_bf16_kernel(const float* __restrict__ in,
                                       bf16_t* __restrict__ out, long n) {
  long i = (long)blockIdx.x * blockDim.x + threadIdx.x;
  long stride = (long)gridDim.x * blockDim.x;
  for (; i < n; i += stride) out[i] = (bf16_t)in[i];
}

// ---------------------------------------------------------------------------
// GEMM:  out[m][n] = sum_k X[m][k] * W[n][k]   (Y = X @ W^T),  N = K = 512.
// Block: 256 threads = 8 waves; block owns a 16-row stripe of X, staged into
// LDS once via the Tensor Data Mover (contiguous 16 KB); each wave computes a
// 16x64 tile.  B fragments are ping-pong double-buffered: the loop is fully
// unrolled (K compile-time), so the two buffers become distinct SSA values
// and no register-rotation moves are generated.
// ---------------------------------------------------------------------------
__global__ __launch_bounds__(256)
void psa_gemm_xwt_kernel(const bf16_t* __restrict__ X, const bf16_t* __restrict__ W,
                         float* __restrict__ outF, bf16_t* __restrict__ outB,
                         const float* __restrict__ bias,
                         const float* __restrict__ residual) {
  constexpr int K = Ec;   // 512
  constexpr int N = Ec;   // 512
  __shared__ __align__(16) bf16_t s_a[16 * K];   // 16 KB A stripe

  const int tid  = threadIdx.x;
  const int lane = tid & 31;
  const int wv   = tid >> 5;
  const int m0   = blockIdx.x * 16;
  const int n0   = wv * 64;                      // 8 waves * 64 = 512 = N

  // ---- stage A stripe global -> LDS ----
#if PSA_HAVE_TDM
  if (wv == 0) {
    psa_tdm_load_to_lds(X + (long)m0 * K, s_a, 16u * K);
    __builtin_amdgcn_s_wait_tensorcnt(0);
  }
#else
  {
    const uint4* src = (const uint4*)(X + (long)m0 * K);
    uint4* dst = (uint4*)s_a;
    for (int i = tid; i < (16 * K) / 8; i += 256) dst[i] = src[i];
  }
#endif
  __syncthreads();

  v8f acc[4];
#pragma unroll
  for (int t = 0; t < 4; ++t)
#pragma unroll
    for (int r = 0; r < 8; ++r) acc[t][r] = 0.0f;

  // ---- K loop: ping-pong B buffers, fully unrolled ----
  v16bf bb[2][4];
  auto loadB = [&](v16bf (&dst)[4], int kkk) {
#pragma unroll
    for (int t = 0; t < 4; ++t)
      dst[t] = load_bT_frag(W + (long)(n0 + t * 16) * K + kkk, K, lane);
  };
  loadB(bb[0], 0);
#pragma unroll
  for (int kk = 0; kk < K; kk += 64) {
    loadB(bb[1], kk + 32);
    {
      v16bf a = load_a_frag(s_a + kk, K, lane);            // LDS
#pragma unroll
      for (int t = 0; t < 4; ++t) acc[t] = wmma_bf16(a, bb[0][t], acc[t]);
    }
    if (kk + 64 < K) loadB(bb[0], kk + 64);                // unroll-resolved
    {
      v16bf a = load_a_frag(s_a + kk + 32, K, lane);       // LDS
#pragma unroll
      for (int t = 0; t < 4; ++t) acc[t] = wmma_bf16(a, bb[1][t], acc[t]);
    }
  }

  // ---- epilogue: C layout col = lane%16, row = r + 8*(lane/16) ----
  const int half = lane >> 4;
  const int col  = lane & 15;
#pragma unroll
  for (int t = 0; t < 4; ++t) {
    const int gn = n0 + t * 16 + col;
#pragma unroll
    for (int r = 0; r < 8; ++r) {
      const int  gm  = m0 + r + 8 * half;
      const long idx = (long)gm * N + gn;
      float v = acc[t][r];
      if (bias)     v += bias[gn];
      if (residual) v += residual[idx];
      if (outF) outF[idx] = v;
      if (outB) outB[idx] = (bf16_t)v;
    }
  }
}

// ---------------------------------------------------------------------------
// Attention kernel: one block per (query-tile of 16, head, batch).
//   Phase 1: scores S = (Qp Kp^T)/8 for 16x2048 via WMMA bf16 -> LDS (128 KB);
//            Q fragments hoisted, K fragments ping-pong buffered (unrolled).
//   Phase 2: per-row max and sum(exp) over full LK (softmax denominator)
//   Phase 3: per-row top-20 scan (softmax monotone => topk of raw scores)
//   Phase 4: out = sum_{j<20} p_j * Vproj[key_j]  (f32 gather), store bf16
// ---------------------------------------------------------------------------
__global__ __launch_bounds__(256)
void psa_attn_kernel(const bf16_t* __restrict__ Qp, const bf16_t* __restrict__ Kp,
                     const float* __restrict__ Vp, bf16_t* __restrict__ attnout) {
  extern __shared__ char smem_raw[];
  float* s_sc     = (float*)smem_raw;              // 16 * 2048
  float* s_red    = s_sc + 16 * LKc;               // 256
  float* s_rowmax = s_red + 256;                   // 16
  float* s_rowsum = s_rowmax + 16;                 // 16
  float* s_topv   = s_rowsum + 16;                 // 16 * 20
  int*   s_topi   = (int*)(s_topv + 16 * TOPKc);   // 16 * 20

  const int qt = blockIdx.x;
  const int h  = blockIdx.y;
  const int b  = blockIdx.z;
  const int tid  = threadIdx.x;
  const int lane = tid & 31;
  const int wv   = tid >> 5;
  const int half = lane >> 4;

  const bf16_t* Qbase = Qp + ((long)(b * LQc + qt * 16)) * Ec + h * HDc;
  const bf16_t* Kbase = Kp + ((long)b * LKc) * Ec + h * HDc;

  // ---- Phase 1: wave wv owns key-tiles [wv*16, wv*16+16) ----
  const v16bf qa0 = load_a_frag(Qbase + 0,  Ec, lane);   // invariant over i
  const v16bf qa1 = load_a_frag(Qbase + 32, Ec, lane);

  const int kt0 = wv * 16;
  v16bf kb[2][2];
  auto loadK = [&](v16bf (&dst)[2], int kt) {
    dst[0] = load_bT_frag(Kbase + (long)(kt * 16) * Ec + 0,  Ec, lane);
    dst[1] = load_bT_frag(Kbase + (long)(kt * 16) * Ec + 32, Ec, lane);
  };
  loadK(kb[0], kt0);
  const int col = lane & 15;
#pragma unroll
  for (int i = 0; i < 16; ++i) {
    const int cur = i & 1;                       // unroll-resolved constant
    if (i < 15) loadK(kb[cur ^ 1], kt0 + i + 1);

    v8f c;
#pragma unroll
    for (int r = 0; r < 8; ++r) c[r] = 0.0f;
    c = wmma_bf16(qa0, kb[cur][0], c);
    c = wmma_bf16(qa1, kb[cur][1], c);

    const int kt = kt0 + i;
#pragma unroll
    for (int r = 0; r < 8; ++r) {
      const int row = r + 8 * half;
      s_sc[row * LKc + kt * 16 + col] = c[r] * 0.125f;   // 1/sqrt(64)
    }
  }
  __syncthreads();

  // ---- Phase 2: row max & sum(exp) (16 threads/row, 128 elems each) ----
  const int row = tid >> 4;
  const int ch  = tid & 15;
  const float* srow = s_sc + row * LKc;
  float lmax = -3.4e38f;
  for (int k = ch * 128; k < ch * 128 + 128; ++k) lmax = fmaxf(lmax, srow[k]);
  s_red[tid] = lmax;
  __syncthreads();
  if (ch == 0) {
    float m = -3.4e38f;
    for (int i = 0; i < 16; ++i) m = fmaxf(m, s_red[row * 16 + i]);
    s_rowmax[row] = m;
  }
  __syncthreads();
  const float rmax = s_rowmax[row];
  float lsum = 0.0f;
  for (int k = ch * 128; k < ch * 128 + 128; ++k) lsum += __expf(srow[k] - rmax);
  s_red[tid] = lsum;
  __syncthreads();
  if (ch == 0) {
    float s = 0.0f;
    for (int i = 0; i < 16; ++i) s += s_red[row * 16 + i];
    s_rowsum[row] = s;
  }
  __syncthreads();

  // ---- Phase 3: top-20 per row (threads 0..15) ----
  if (tid < 16) {
    const float* sr = s_sc + tid * LKc;
    float tv[TOPKc];
    int   ti[TOPKc];
#pragma unroll
    for (int j = 0; j < TOPKc; ++j) { tv[j] = -3.4e38f; ti[j] = 0; }
    for (int k = 0; k < LKc; ++k) {
      const float s = sr[k];
      if (s > tv[TOPKc - 1]) {
        int p = TOPKc - 1;
        while (p > 0 && tv[p - 1] < s) {
          tv[p] = tv[p - 1]; ti[p] = ti[p - 1]; --p;
        }
        tv[p] = s; ti[p] = k;
      }
    }
    const float m   = s_rowmax[tid];
    const float inv = 1.0f / s_rowsum[tid];
#pragma unroll
    for (int j = 0; j < TOPKc; ++j) {
      s_topv[tid * TOPKc + j] = __expf(tv[j] - m) * inv;  // softmax prob
      s_topi[tid * TOPKc + j] = ti[j];
    }
  }
  __syncthreads();

  // ---- Phase 4: sparse gather of V (16 rows x 64 dims, 4 dims/thread) ----
  const int orow = tid >> 4;
  const int d0   = (tid & 15) * 4;
  float4 acc = make_float4(0.f, 0.f, 0.f, 0.f);
  for (int j = 0; j < TOPKc; ++j) {
    const int   key = s_topi[orow * TOPKc + j];
    const float p   = s_topv[orow * TOPKc + j];
    const float4 vv = *(const float4*)(Vp + ((long)(b * LKc + key)) * Ec + h * HDc + d0);
    acc.x += p * vv.x; acc.y += p * vv.y; acc.z += p * vv.z; acc.w += p * vv.w;
  }
  bf16_t* outp = attnout + ((long)(b * LQc + qt * 16 + orow)) * Ec + h * HDc + d0;
  outp[0] = (bf16_t)acc.x; outp[1] = (bf16_t)acc.y;
  outp[2] = (bf16_t)acc.z; outp[3] = (bf16_t)acc.w;
}

// ---------------------------------------------------------------------------
// Host-side launch
// ---------------------------------------------------------------------------
extern "C" void kernel_launch(void* const* d_in, const int* in_sizes, int n_in,
                              void* d_out, int out_size, void* d_ws, size_t ws_size,
                              hipStream_t stream) {
  (void)in_sizes; (void)n_in; (void)out_size; (void)ws_size;

  const float* q  = (const float*)d_in[0];   // [B,LQ,E]
  const float* k  = (const float*)d_in[1];   // [B,LK,E]
  const float* v  = (const float*)d_in[2];   // [B,LK,E]
  const float* Wq = (const float*)d_in[3];   // [E,E]
  const float* Wk = (const float*)d_in[4];
  const float* Wv = (const float*)d_in[5];
  const float* Wo = (const float*)d_in[6];
  const float* bo = (const float*)d_in[7];   // [E]
  float* out = (float*)d_out;                // [B,LQ,E]

  const long nW  = (long)Ec * Ec;            //   262144
  const long nQ  = (long)Bc * LQc * Ec;      //  2097152
  const long nKV = (long)Bc * LKc * Ec;      //  8388608

  // carve workspace (256B aligned blocks)
  char* p = (char*)d_ws;
  auto carve = [&](size_t bytes) -> void* {
    void* r = (void*)p;
    p += (bytes + 255) & ~(size_t)255;
    return r;
  };
  bf16_t* Wqb     = (bf16_t*)carve(nW  * sizeof(bf16_t));
  bf16_t* Wkb     = (bf16_t*)carve(nW  * sizeof(bf16_t));
  bf16_t* Wvb     = (bf16_t*)carve(nW  * sizeof(bf16_t));
  bf16_t* Wob     = (bf16_t*)carve(nW  * sizeof(bf16_t));
  bf16_t* qb      = (bf16_t*)carve(nQ  * sizeof(bf16_t));
  bf16_t* kb      = (bf16_t*)carve(nKV * sizeof(bf16_t));
  bf16_t* vb      = (bf16_t*)carve(nKV * sizeof(bf16_t));
  bf16_t* Qp      = (bf16_t*)carve(nQ  * sizeof(bf16_t));
  bf16_t* Kp      = (bf16_t*)carve(nKV * sizeof(bf16_t));
  float*  Vp      = (float*) carve(nKV * sizeof(float));
  bf16_t* attnout = (bf16_t*)carve(nQ  * sizeof(bf16_t));

  // 1) bf16 conversions
  psa_f32_to_bf16_kernel<<<512,  256, 0, stream>>>(Wq, Wqb, nW);
  psa_f32_to_bf16_kernel<<<512,  256, 0, stream>>>(Wk, Wkb, nW);
  psa_f32_to_bf16_kernel<<<512,  256, 0, stream>>>(Wv, Wvb, nW);
  psa_f32_to_bf16_kernel<<<512,  256, 0, stream>>>(Wo, Wob, nW);
  psa_f32_to_bf16_kernel<<<2048, 256, 0, stream>>>(q,  qb,  nQ);
  psa_f32_to_bf16_kernel<<<4096, 256, 0, stream>>>(k,  kb,  nKV);
  psa_f32_to_bf16_kernel<<<4096, 256, 0, stream>>>(v,  vb,  nKV);

  // 2) projections (Y = X @ W^T) via WMMA
  const int MQ = Bc * LQc;   // 4096
  const int MK = Bc * LKc;   // 16384
  psa_gemm_xwt_kernel<<<dim3(MQ / 16), 256, 0, stream>>>(
      qb, Wqb, nullptr, Qp, nullptr, nullptr);
  psa_gemm_xwt_kernel<<<dim3(MK / 16), 256, 0, stream>>>(
      kb, Wkb, nullptr, Kp, nullptr, nullptr);
  psa_gemm_xwt_kernel<<<dim3(MK / 16), 256, 0, stream>>>(
      vb, Wvb, Vp, nullptr, nullptr, nullptr);

  // 3) attention: scores -> softmax stats -> top-20 -> sparse V gather
  const size_t smem = (size_t)(16 * LKc + 256 + 16 + 16 + 16 * TOPKc) * sizeof(float)
                      + (size_t)(16 * TOPKc) * sizeof(int);
  psa_attn_kernel<<<dim3(LQc / 16, Hc, Bc), 256, smem, stream>>>(Qp, Kp, Vp, attnout);

  // 4) output projection + bias + residual  -> f32 d_out
  psa_gemm_xwt_kernel<<<dim3(MQ / 16), 256, 0, stream>>>(
      attnout, Wob, out, nullptr, bo, q);
}
